// Model_57440892617422
// MI455X (gfx1250) — matrix operand order repaired and checked by reading, not compiled, not run on hardware
//
#include <hip/hip_runtime.h>
#include <hip/hip_bf16.h>
#include <math.h>

// Problem constants (from reference)
#define B_    64
#define SC    640
#define SN    128
#define D_    768
#define NROLE 12

typedef __attribute__((ext_vector_type(16))) __bf16 v16bf;
typedef __attribute__((ext_vector_type(8)))  __bf16 v8bf;
typedef __attribute__((ext_vector_type(8)))  float  v8f;

// ---- output layout (tuple concatenated, f32) ----
#define CODE_H_OFF  ((size_t)0)
#define CODE_P_OFF  (CODE_H_OFF + (size_t)B_*SC*D_)   // 31,457,280
#define NL_H_OFF    (CODE_P_OFF + (size_t)B_*SC)      // 31,498,240
#define NL_P_OFF    (NL_H_OFF  + (size_t)B_*SN*D_)    // 37,789,696
#define SIM_OFF     (NL_P_OFF  + (size_t)B_*SN)       // 37,797,888

// ---- workspace layout (4-byte element indices); ~165 KB used ----
#define WS_INV_OFF  0                   // float inv_count[B*SC]
#define WS_CT_OFF   (B_*SC)             // int   codeTotal[B]
#define WS_NT_OFF   (WS_CT_OFF + B_)    // int   nlTotal[B]
#define WS_RD_OFF   (WS_NT_OFF + B_)    // float roleDot[NROLE]

// =====================================================================
// k_prep: per-batch valid lengths (first eos==2) and roleDot[r]
// =====================================================================
__global__ void __launch_bounds__(256)
k_prep(const int* __restrict__ code_inputs, const int* __restrict__ nl_inputs,
       const float* __restrict__ role_table, const float* __restrict__ w_code,
       float* __restrict__ ws)
{
  int tid = threadIdx.x;
  int* iws = (int*)ws;
  if (tid < B_) {
    int b = tid; int t = 254;                          // CODE_DEFAULT
    for (int s = 0; s < SC; ++s) { if (code_inputs[b*SC + s] == 2) { t = s; break; } }
    iws[WS_CT_OFF + b] = t;
  } else if (tid < 2*B_) {
    int b = tid - B_; int t = 126;                     // NL_DEFAULT
    for (int s = 0; s < SN; ++s) { if (nl_inputs[b*SN + s] == 2) { t = s; break; } }
    iws[WS_NT_OFF + b] = t;
  } else if (tid < 2*B_ + NROLE) {
    int r = tid - 2*B_; float acc = 0.f;
    for (int d = 0; d < D_; ++d) acc += role_table[r*D_ + d] * w_code[d];
    ws[WS_RD_OFF + r] = acc;
  }
}

// =====================================================================
// k_counts: one wave per (b,n) row -> 1/(rowcount + 1e-10) for node rows
// =====================================================================
__global__ void __launch_bounds__(256)
k_counts(const int* __restrict__ position_idx, const unsigned char* __restrict__ attn,
         float* __restrict__ ws)
{
  int lane = threadIdx.x & 31;
  int w    = threadIdx.x >> 5;
  int row  = blockIdx.x * 8 + w;                       // row = b*SC + n
  if (position_idx[row] != 0) {                        // not a DFG node
    if (lane == 0) ws[WS_INV_OFF + row] = 0.f;
    return;
  }
  const unsigned char* arow = attn + (size_t)row * SC;
  const int* prow = position_idx + (row / SC) * SC;    // position row of batch b
  int cnt = 0;
  for (int t = lane; t < SC; t += 32)
    cnt += ((prow[t] >= 2) && arow[t]) ? 1 : 0;
  #pragma unroll
  for (int m = 16; m >= 1; m >>= 1) cnt += __shfl_xor(cnt, m, 32);
  if (lane == 0) ws[WS_INV_OFF + row] = 1.0f / ((float)cnt + 1e-10f);
}

// =====================================================================
// k_nl: one wave per nl row: gather + l2norm + probs
// =====================================================================
__global__ void __launch_bounds__(256)
k_nl(const int* __restrict__ nl_inputs, const float* __restrict__ W_emb,
     const float* __restrict__ w_nl, const float* __restrict__ b_nl,
     const float* __restrict__ ws, float* __restrict__ out)
{
  int lane = threadIdx.x & 31;
  int w    = threadIdx.x >> 5;
  int row  = blockIdx.x * 8 + w;                       // 0..B*SN-1
  int b = row >> 7, s = row & 127;
  int tok = nl_inputs[row];
  const float* base = W_emb + (size_t)tok * D_;
  float v[24]; float sq = 0.f, dw = 0.f;
  #pragma unroll
  for (int j = 0; j < 24; ++j) {
    float x = base[lane + 32*j];
    v[j] = x; sq += x*x; dw += x * w_nl[lane + 32*j];
  }
  #pragma unroll
  for (int m = 16; m >= 1; m >>= 1) { sq += __shfl_xor(sq, m, 32); dw += __shfl_xor(dw, m, 32); }
  float den = fmaxf(sqrtf(sq), 1e-12f);
  float* oh = out + NL_H_OFF + (size_t)row * D_;
  #pragma unroll
  for (int j = 0; j < 24; ++j) oh[lane + 32*j] = v[j] / den;
  if (lane == 0) {
    const int* iws = (const int*)ws;
    float x = dw / den + b_nl[0];
    float p = 1.f / (1.f + expf(-x));
    out[NL_P_OFF + row] = p * ((s < iws[WS_NT_OFF + b]) ? 1.f : 0.f);
  }
}

// =====================================================================
// k_code: WMMA mask-GEMM + fused mixing, l2norm, probs.
// Grid (Sc/16, B), 256 threads (8 waves x 6 N-tiles = 768 cols).
// =====================================================================
#define KPA 40   // padded A row (bf16), 80B stride keeps 16B alignment
__global__ void __launch_bounds__(256)
k_code(const int* __restrict__ code_inputs, const unsigned char* __restrict__ attn,
       const int* __restrict__ position_idx, const int* __restrict__ role_indices,
       const float* __restrict__ W_emb, const float* __restrict__ b_code,
       const float* __restrict__ w_code, const float* __restrict__ ws,
       float* __restrict__ out)
{
  __shared__ __bf16 sA[16 * KPA];            // mask tile, fragment-ready
  __shared__ __bf16 sBT[D_ * 32];            // emb tile transposed: [col][k]
  __shared__ float  swc[D_];
  __shared__ unsigned long long sOff[32];    // W_emb byte offsets of K-step rows
  __shared__ unsigned char sTokMask[SC];     // token_mask (pos>=2) for whole batch row
  __shared__ int    sRowTok[16];
  __shared__ int    sNode[16];
  __shared__ float  sInv[16];
  __shared__ float  redSq[8 * 16], redDw[8 * 16];
  __shared__ float  sDen[16];

  const int tid  = threadIdx.x;
  const int lane = tid & 31, w = tid >> 5;
  const int lo   = lane & 15, hi = lane >> 4;
  const int b    = blockIdx.y;
  const int n0   = blockIdx.x * 16;

  for (int e = tid; e < D_; e += 256) swc[e] = w_code[e];
  for (int e = tid; e < SC; e += 256)
    sTokMask[e] = (position_idx[b*SC + e] >= 2) ? 1 : 0;
  if (tid < 16) {
    int n = n0 + tid;
    sRowTok[tid] = code_inputs[b*SC + n];
    sNode[tid]   = (position_idx[b*SC + n] == 0) ? 1 : 0;
    sInv[tid]    = ws[WS_INV_OFF + b*SC + n];
  }
  __syncthreads();

  v8f acc[6] = {};
  const char* wb = (const char*)W_emb;

  for (int k0 = 0; k0 < SC; k0 += 32) {
    __syncthreads();                                   // prior-iter LDS consumers done
    if (tid < 32)
      sOff[tid] = (unsigned long long)code_inputs[b*SC + k0 + tid] * (D_ * 4);
    __syncthreads();
    // A tile: binary mask (exact in bf16); 512 entries, 2 per thread
    for (int e = tid; e < 512; e += 256) {
      int r = e >> 5, k = e & 31;
      int tg = k0 + k;
      int v = sNode[r] && sTokMask[tg] &&
              attn[((size_t)(b*SC + n0 + r)) * SC + tg];
      sA[r*KPA + k] = (__bf16)(float)v;
    }
    // B tile transposed: each thread packs an 8-deep column segment -> b128 store
    for (int e = tid; e < D_ * 4; e += 256) {
      int col = e % D_, tg = e / D_;                   // tg = 0..3
      int t0 = tg * 8;
      v8bf pk;
      #pragma unroll
      for (int j = 0; j < 8; ++j)
        pk[j] = (__bf16)(*(const float*)(wb + sOff[t0 + j] + (size_t)col * 4));
      *(v8bf*)&sBT[col*32 + t0] = pk;
    }
    __syncthreads();

    // A fragment: row = lo; elems 0..7 -> K=8*hi+0..7, 8..15 -> K=16+8*hi+0..7
    v16bf afrag;
    {
      const __bf16* base = &sA[lo*KPA];
      v8bf x = *(const v8bf*)(base + 8*hi);
      v8bf y = *(const v8bf*)(base + 16 + 8*hi);
      #pragma unroll
      for (int i = 0; i < 8; ++i) { afrag[i] = x[i]; afrag[8+i] = y[i]; }
    }
    // load ALL six B fragments first, then a burst of WMMAs (fewer ds waits)
    v16bf bfrag[6];
    #pragma unroll
    for (int j = 0; j < 6; ++j) {
      int col = (w*6 + j)*16 + lo;                     // B: K = 8*hi + 0..7, 16+8*hi + 0..7
      const __bf16* base = &sBT[col*32];
      v8bf x = *(const v8bf*)(base + 8*hi);
      v8bf y = *(const v8bf*)(base + 16 + 8*hi);
      #pragma unroll
      for (int i = 0; i < 8; ++i) { bfrag[j][i] = x[i]; bfrag[j][8+i] = y[i]; }
    }
    #pragma unroll
    for (int j = 0; j < 6; ++j)
      acc[j] = __builtin_amdgcn_wmma_f32_16x16x32_bf16(
                 false, afrag, false, bfrag[j], (short)0, acc[j], false, false);
  }
  __syncthreads();

  // Epilogue: scale node rows, substitute emb for non-node rows, reduce
  float psq[8], pdw[8];
  #pragma unroll
  for (int r = 0; r < 8; ++r) { psq[r] = 0.f; pdw[r] = 0.f; }
  #pragma unroll
  for (int j = 0; j < 6; ++j) {
    int col = (w*6 + j)*16 + lo;
    #pragma unroll
    for (int r = 0; r < 8; ++r) {
      int M = r + 8*hi;                                // D layout: VGPR r, lane-half hi
      float val = acc[j][r];
      if (sNode[M]) val *= sInv[M];
      else          val  = W_emb[(size_t)sRowTok[M] * D_ + col];
      acc[j][r] = val;
      psq[r] += val * val;
      pdw[r] += val * swc[col];
    }
  }
  #pragma unroll
  for (int r = 0; r < 8; ++r) {
    #pragma unroll
    for (int m = 8; m >= 1; m >>= 1) {
      psq[r] += __shfl_xor(psq[r], m, 32);
      pdw[r] += __shfl_xor(pdw[r], m, 32);
    }
  }
  if (lo == 0) {
    #pragma unroll
    for (int r = 0; r < 8; ++r) {
      int M = r + 8*hi;
      redSq[w*16 + M] = psq[r];
      redDw[w*16 + M] = pdw[r];
    }
  }
  __syncthreads();
  if (tid < 16) {
    float sq = 0.f, dw = 0.f;
    #pragma unroll
    for (int ww = 0; ww < 8; ++ww) { sq += redSq[ww*16 + tid]; dw += redDw[ww*16 + tid]; }
    float den = fmaxf(sqrtf(sq), 1e-12f);
    sDen[tid] = den;
    int n = n0 + tid;
    const int* iws = (const int*)ws;
    int role = role_indices[b*SC + n];
    float x = dw / den + ws[WS_RD_OFF + role] + b_code[0];
    float p = 1.f / (1.f + expf(-x));
    out[CODE_P_OFF + b*SC + n] = p * ((n < iws[WS_CT_OFF + b]) ? 1.f : 0.f);
  }
  __syncthreads();
  #pragma unroll
  for (int j = 0; j < 6; ++j) {
    int col = (w*6 + j)*16 + lo;
    #pragma unroll
    for (int r = 0; r < 8; ++r) {
      int M = r + 8*hi;
      out[CODE_H_OFF + ((size_t)(b*SC + n0 + M)) * D_ + col] = acc[j][r] / sDen[M];
    }
  }
}

// =====================================================================
// k_sim: sim = nl_hidden @ code_hidden^T via WMMA (bf16 in, f32 acc).
// Both fragments are per-lane contiguous rows -> direct float4 loads.
// Grid (Sc/128, Sn/16, B), 256 threads (8 waves x one 16-col N-tile).
// =====================================================================
__device__ __forceinline__ void cvt4(v16bf& f, int base, float4 a) {
  f[base+0] = (__bf16)a.x; f[base+1] = (__bf16)a.y;
  f[base+2] = (__bf16)a.z; f[base+3] = (__bf16)a.w;
}

__global__ void __launch_bounds__(256)
k_sim(float* __restrict__ out)
{
  const int tid  = threadIdx.x;
  const int lane = tid & 31, w = tid >> 5;
  const int lo   = lane & 15, hi = lane >> 4;
  const int b  = blockIdx.z;
  const int m0 = blockIdx.y * 16;
  const int n0 = (blockIdx.x * 8 + w) * 16;

  const float* Arow = out + NL_H_OFF   + ((size_t)(b*SN + m0 + lo)) * D_;
  const float* Brow = out + CODE_H_OFF + ((size_t)(b*SC + n0 + lo)) * D_;

  v8f acc = {};
  for (int k0 = 0; k0 < D_; k0 += 32) {
    v16bf af, bfv;
    {
      const float4* p = (const float4*)(Arow + k0 + 8*hi);       // K = 8*hi + 0..7
      const float4* q = (const float4*)(Arow + k0 + 16 + 8*hi);  // K = 16+8*hi + 0..7
      cvt4(af, 0, p[0]); cvt4(af, 4, p[1]);
      cvt4(af, 8, q[0]); cvt4(af, 12, q[1]);
    }
    {
      const float4* p = (const float4*)(Brow + k0 + 8*hi);       // K = 8*hi + 0..7
      const float4* q = (const float4*)(Brow + k0 + 16 + 8*hi);  // K = 16+8*hi + 0..7
      cvt4(bfv, 0, p[0]); cvt4(bfv, 4, p[1]);
      cvt4(bfv, 8, q[0]); cvt4(bfv, 12, q[1]);
    }
    acc = __builtin_amdgcn_wmma_f32_16x16x32_bf16(
            false, af, false, bfv, (short)0, acc, false, false);
  }
  #pragma unroll
  for (int r = 0; r < 8; ++r) {
    int M = m0 + r + 8*hi;
    int N = n0 + lo;
    out[SIM_OFF + ((size_t)(b*SN + M)) * SC + N] = acc[r];
  }
}

// =====================================================================
extern "C" void kernel_launch(void* const* d_in, const int* in_sizes, int n_in,
                              void* d_out, int out_size, void* d_ws, size_t ws_size,
                              hipStream_t stream) {
  const int*           code_inputs  = (const int*)d_in[0];
  const unsigned char* attn         = (const unsigned char*)d_in[1]; // jax bool -> 1 byte
  const int*           position_idx = (const int*)d_in[2];
  const int*           nl_inputs    = (const int*)d_in[3];
  const int*           role_indices = (const int*)d_in[4];
  const float*         W_emb        = (const float*)d_in[5];
  const float*         role_table   = (const float*)d_in[6];
  const float*         w_code       = (const float*)d_in[7];
  const float*         b_code       = (const float*)d_in[8];
  const float*         w_nl         = (const float*)d_in[9];
  const float*         b_nl         = (const float*)d_in[10];
  float* out = (float*)d_out;
  float* ws  = (float*)d_ws;   // uses ~165 KB

  k_prep  <<<1, 256, 0, stream>>>(code_inputs, nl_inputs, role_table, w_code, ws);
  k_counts<<<(B_*SC)/8, 256, 0, stream>>>(position_idx, attn, ws);
  k_nl    <<<(B_*SN)/8, 256, 0, stream>>>(nl_inputs, W_emb, w_nl, b_nl, ws, out);
  k_code  <<<dim3(SC/16, B_), 256, 0, stream>>>(code_inputs, attn, position_idx,
                                                role_indices, W_emb, b_code, w_code,
                                                ws, out);
  k_sim   <<<dim3(SC/128, SN/16, B_), 256, 0, stream>>>(out);
}
